// GeoDavidBlockCompanion_51797305589994
// MI455X (gfx1250) — compile-verified, matrix operand order checked by reading, not run on hardware
//
#include <hip/hip_runtime.h>
#include <hip/hip_bf16.h>
#include <cmath>

// ---------------------------------------------------------------------------
// Problem constants (from the reference)
// ---------------------------------------------------------------------------
#define BATCH   16384
#define IN_DIM  1280
#define BELLY   1024
#define SCALE   512
#define NBINS   100
#define NPAT    10
#define LEVELS  12

// d_out layout (floats, concatenated in return order)
#define OFF_Z   0
#define OFF_TL  (BATCH * SCALE)                       // 8,388,608
#define OFF_PL  (OFF_TL + BATCH * NBINS)              // 10,027,008
#define OFF_TC  (OFF_PL + BATCH * NBINS * NPAT)       // 26,411,008
#define OFF_CV  (OFF_TC + BATCH)                      // 26,427,392

// swizzled-weight tile counts (16-col tiles, padded to multiples of 8 = BN/16)
#define NT_W1   (BELLY / 16)                          // 64
#define NT_W2   (SCALE / 16)                          // 32
#define NT_WT   8                                     // ceil(100/128)*8
#define NT_WP   64                                    // ceil(1000/128)*8

// workspace layout (bytes)
#define WS_H16     0                                          // BATCH*BELLY f16
#define WS_Z16     (BATCH * BELLY * 2)                        // BATCH*SCALE f16
#define WS_W1S     (WS_Z16 + BATCH * SCALE * 2)
#define WS_W2S     (WS_W1S + (IN_DIM / 32) * NT_W1 * 1024)
#define WS_WTS     (WS_W2S + (BELLY / 32) * NT_W2 * 1024)
#define WS_WPS     (WS_WTS + (SCALE / 32) * NT_WT * 1024)
#define WS_DOT     (WS_WPS + (SCALE / 32) * NT_WP * 1024)
#define WS_MAXPOS  (WS_DOT + BATCH * 4)

typedef __attribute__((ext_vector_type(16))) _Float16 v16h;
typedef __attribute__((ext_vector_type(8)))  _Float16 v8h;
typedef __attribute__((ext_vector_type(8)))  float    v8f;

struct half4 { _Float16 x, y, z, w; };

// ---------------------------------------------------------------------------
// Weight pre-swizzle: W[K,N] f32 row-major -> Wf in WMMA B-fragment layout:
//   Wf[((kb*NT + nt)*32 + lane)*16 + j] = W[kb*32 + kk][nt*16 + (lane&15)]
//   kk = (j&7) + ((j>>3)<<4) + (lane>>4)*8      (ISA 7.12.2, 16-bit B 32x16)
// Columns >= N are zero-filled so the GEMM hot loop needs no N guards.
// ---------------------------------------------------------------------------
__global__ __launch_bounds__(256)
void swizzle_w_kernel(const float* __restrict__ W, _Float16* __restrict__ Wf,
                      int K, int N, int NT)
{
    int t = blockIdx.x * 256 + threadIdx.x;       // one thread per (kb, nt, lane)
    int total = (K >> 5) * NT * 32;
    if (t >= total) return;
    int lane   = t & 31;
    int nt     = (t >> 5) % NT;
    int kb     = (t >> 5) / NT;
    int laneLo = lane & 15;
    int laneHi = lane >> 4;
    int col    = nt * 16 + laneLo;

    v16h v;
    #pragma unroll
    for (int j = 0; j < 16; ++j) {
        int kk   = (j & 7) + ((j >> 3) << 4) + laneHi * 8;
        int krow = kb * 32 + kk;
        v[j] = (col < N) ? (_Float16)W[(size_t)krow * N + col] : (_Float16)0.f;
    }
    *(v16h*)(Wf + ((size_t)t << 4)) = v;
}

// ---------------------------------------------------------------------------
// WMMA GEMM:  C[M,N] = act(A[M,K] @ W[K,N] + bias), W pre-swizzled (Wf).
//   256 threads = 8 waves; tile BM=128 x BN=128, k-step 32.
//   B panel staged by async global->LDS DMA (double-buffered, ASYNCcnt);
//   A tile staged through registers (f32 path converts to f16 on the way).
//   Wave (waveM 0..3, waveN 0..1) owns 32x64 = 2x4 wmma 16x16x32 tiles.
// ---------------------------------------------------------------------------
#define BM 128
#define BN 128
#define LDA 40   // 32 + 8 halfs pad (80B rows keep 16B alignment)

template<int ACT, bool A_HALF, bool OUT_HALF>
__global__ __launch_bounds__(256)
void gemm_wmma_kernel(const void* __restrict__ Ap, const _Float16* __restrict__ Wf,
                      const float* __restrict__ bias, void* __restrict__ Cp,
                      int M, int N, int NT, int K)
{
    __shared__ __align__(32) _Float16 sA[BM * LDA];      // row-major A tile (10 KB)
    __shared__ __align__(32) _Float16 sB[2][8 * 512];    // frag layout, dbl-buf (16 KB)

    const int tid    = threadIdx.x;
    const int lane   = tid & 31;
    const int wave   = tid >> 5;
    const int waveM  = wave >> 1;          // 0..3  (32 rows each)
    const int waveN  = wave & 1;           // 0..1  (64 cols each)
    const int laneLo = lane & 15;
    const int laneHi = lane >> 4;

    const int rowBase = blockIdx.y * BM;
    const int colBase = blockIdx.x * BN;
    const int ct0     = blockIdx.x * 8;    // first 16-col tile of this block
    const int K32     = K >> 5;

    v8f acc[2][4] = {};

    float4 areg[A_HALF ? 2 : 4];

    // ---- async global->LDS copy of the 8KB B panel for step kb (ASYNCcnt) ----
    auto async_loadB = [&](int kb) {
        const _Float16* bsrc = Wf + (((size_t)(kb * NT + ct0)) << 9) + (tid << 3);
        unsigned lds0 = (unsigned)(uintptr_t)&sB[kb & 1][tid << 3];
        asm volatile("global_load_async_to_lds_b128 %0, %1, off"
                     :: "v"(lds0), "v"(bsrc) : "memory");
        asm volatile("global_load_async_to_lds_b128 %0, %1, off"
                     :: "v"(lds0 + 4096u), "v"(bsrc + 2048) : "memory");
    };

    auto load_tilesA = [&](int kb) {
        if constexpr (A_HALF) {
            #pragma unroll
            for (int i = 0; i < 2; ++i) {
                int g = tid + (i << 8);            // 0..511 : 8-half chunks
                int r = g >> 2;                    // 4 chunks per 32-half row
                int c = (g & 3) << 3;
                areg[i] = *(const float4*)((const _Float16*)Ap +
                            (size_t)(rowBase + r) * K + (kb << 5) + c);
            }
        } else {
            #pragma unroll
            for (int i = 0; i < 4; ++i) {
                int g = tid + (i << 8);            // 0..1023 : 4-float chunks
                int r = g >> 3;                    // 8 chunks per 32-float row
                int c = (g & 7) << 2;
                areg[i] = *(const float4*)((const float*)Ap +
                            (size_t)(rowBase + r) * K + (kb << 5) + c);
            }
        }
    };

    async_loadB(0);
    load_tilesA(0);

    for (int kb = 0; kb < K32; ++kb) {
        // this wave's async B(kb) writes have landed in LDS
        asm volatile("s_wait_asynccnt 0x0" ::: "memory");
        __syncthreads();   // all waves' B(kb) landed + prev fragment reads done

        // ---- stage A tile from registers: all 16B / 8B stores ----
        if constexpr (A_HALF) {
            #pragma unroll
            for (int i = 0; i < 2; ++i) {
                int g = tid + (i << 8);
                int r = g >> 2;
                int c = (g & 3) << 3;
                *(float4*)&sA[r * LDA + c] = areg[i];
            }
        } else {
            #pragma unroll
            for (int i = 0; i < 4; ++i) {
                int g = tid + (i << 8);
                int r = g >> 3;
                int c = (g & 7) << 2;
                half4 hv;
                hv.x = (_Float16)areg[i].x; hv.y = (_Float16)areg[i].y;
                hv.z = (_Float16)areg[i].z; hv.w = (_Float16)areg[i].w;
                *(half4*)&sA[r * LDA + c] = hv;
            }
        }

        // ---- issue next step's loads; latency hidden behind the WMMAs ----
        if (kb + 1 < K32) {
            async_loadB(kb + 1);       // other sB buffer: no reader conflict
            load_tilesA(kb + 1);
        }

        __syncthreads();

        // ---- fragments: pure b128 LDS loads, zero packing ----
        const _Float16* sBk = sB[kb & 1];
        v16h afrag[2], bfrag[4];
        #pragma unroll
        for (int tm = 0; tm < 2; ++tm) {
            const _Float16* ap = &sA[(waveM * 32 + tm * 16 + laneLo) * LDA + laneHi * 8];
            v8h lo = *(const v8h*)ap;
            v8h hi = *(const v8h*)(ap + 16);
            afrag[tm] = __builtin_shufflevector(lo, hi,
                0, 1, 2, 3, 4, 5, 6, 7, 8, 9, 10, 11, 12, 13, 14, 15);
        }
        #pragma unroll
        for (int tn = 0; tn < 4; ++tn)
            bfrag[tn] = *(const v16h*)&sBk[((waveN * 4 + tn) << 9) + (lane << 4)];

        #pragma unroll
        for (int tm = 0; tm < 2; ++tm)
            #pragma unroll
            for (int tn = 0; tn < 4; ++tn)
                acc[tm][tn] = __builtin_amdgcn_wmma_f32_16x16x32_f16(
                    false, afrag[tm], false, bfrag[tn],
                    (short)0, acc[tm][tn], false, false);
    }

    // ---- epilogue: C/D layout -> M = r + laneHi*8, N = laneLo ----
    #pragma unroll
    for (int tm = 0; tm < 2; ++tm) {
        #pragma unroll
        for (int tn = 0; tn < 4; ++tn) {
            int gn = colBase + waveN * 64 + tn * 16 + laneLo;
            if (gn >= N) continue;
            float bv = bias ? bias[gn] : 0.f;
            #pragma unroll
            for (int r = 0; r < 8; ++r) {
                int gm = rowBase + waveM * 32 + tm * 16 + laneHi * 8 + r;
                float val = acc[tm][tn][r] + bv;
                if (ACT == 1) val = fmaxf(val, 0.f);
                if constexpr (OUT_HALF)
                    ((_Float16*)Cp)[(size_t)gm * N + gn] = (_Float16)val;
                else
                    ((float*)Cp)[(size_t)gm * N + gn] = val;
            }
        }
    }
}

// ---------------------------------------------------------------------------
// Row L2-normalization (one wave per row) fused with dot(z_norm, w_pos).
// ---------------------------------------------------------------------------
__global__ __launch_bounds__(256)
void normalize_kernel(float* __restrict__ z, _Float16* __restrict__ zh,
                      float* __restrict__ dotpos, const float* __restrict__ w_pos)
{
    int row  = blockIdx.x * 8 + (threadIdx.x >> 5);
    int lane = threadIdx.x & 31;
    if (row >= BATCH) return;
    float* zr = z + (size_t)row * SCALE;

    float ss = 0.f, dp = 0.f;
    for (int i = lane; i < SCALE; i += 32) {
        float v = zr[i];
        ss += v * v;
        dp += v * w_pos[i];
    }
    #pragma unroll
    for (int off = 16; off; off >>= 1) {
        ss += __shfl_down(ss, off, 32);
        dp += __shfl_down(dp, off, 32);
    }
    ss = __shfl(ss, 0, 32);
    dp = __shfl(dp, 0, 32);

    float inv = 1.f / fmaxf(sqrtf(ss), 1e-12f);
    for (int i = lane; i < SCALE; i += 32) {
        float v = zr[i] * inv;
        zr[i] = v;
        zh[(size_t)row * SCALE + i] = (_Float16)v;
    }
    if (lane == 0) dotpos[row] = dp * inv;
}

// ---------------------------------------------------------------------------
// timestep_class + global max_pos reduction
// ---------------------------------------------------------------------------
__global__ void zero_maxpos_kernel(int* maxpos) { maxpos[0] = 0; }

__global__ __launch_bounds__(256)
void tclass_kernel(const int* __restrict__ t, float* __restrict__ out_tc,
                   int* __restrict__ maxpos)
{
    int i = blockIdx.x * 256 + threadIdx.x;
    if (i >= BATCH) return;
    int tc = (int)((float)t[i] / 1000.0f * (float)NBINS);
    tc = min(max(tc, 0), NBINS - 1);
    out_tc[i] = (float)tc;
    atomicMax(maxpos, tc);
}

// ---------------------------------------------------------------------------
// Cantor staircase (elementwise; shift dot precomputed by normalize_kernel)
// ---------------------------------------------------------------------------
__global__ __launch_bounds__(256)
void cantor_kernel(const int* __restrict__ t, const float* __restrict__ dotpos,
                   const float* __restrict__ b_pos, const float* __restrict__ alpha_p,
                   const int* __restrict__ maxpos, float* __restrict__ out)
{
    int i = blockIdx.x * 256 + threadIdx.x;
    if (i >= BATCH) return;

    int tc = (int)((float)t[i] / 1000.0f * (float)NBINS);
    tc = min(max(tc, 0), NBINS - 1);

    int   max_pos = maxpos[0] + 1;
    float denom   = fmaxf((float)(max_pos - 1), 1.f);
    float pos_f   = (float)tc;
    float x       = (max_pos > 1) ? (pos_f / denom) : pos_f;
    x = fminf(fmaxf(x, 1e-6f), 1.f - 1e-6f);

    float shift = tanhf(dotpos[i] + b_pos[0]) * 0.3f;
    x = fminf(fmaxf(x + shift, 1e-6f), 1.f - 1e-6f);

    const float alpha = alpha_p[0];
    const float itau  = 1.f / (0.25f + 1e-8f);
    float Cx = 0.f, w = 0.5f;
    #pragma unroll
    for (int l = 0; l < LEVELS; ++l) {
        float y  = x * 3.0f;
        float d0 = y - 0.5f, d1 = y - 1.5f, d2 = y - 2.5f;
        float l0 = -d0 * d0 * itau;
        float l1 = -d1 * d1 * itau;
        float l2 = -d2 * d2 * itau;
        float m  = fmaxf(l0, fmaxf(l1, l2));
        float e0 = expf(l0 - m), e1 = expf(l1 - m), e2 = expf(l2 - m);
        float bit = (e1 * alpha + e2) / (e0 + e1 + e2);
        Cx += bit * w;
        x = y - floorf(y);
        w *= 0.5f;
    }
    out[i] = fminf(fmaxf(Cx, 0.f), 1.f);
}

// ---------------------------------------------------------------------------
// Host launcher
// ---------------------------------------------------------------------------
extern "C" void kernel_launch(void* const* d_in, const int* in_sizes, int n_in,
                              void* d_out, int out_size, void* d_ws, size_t ws_size,
                              hipStream_t stream)
{
    (void)in_sizes; (void)n_in; (void)out_size; (void)ws_size;

    const float* features  = (const float*)d_in[0];
    const int*   timesteps = (const int*)  d_in[1];
    const float* W1        = (const float*)d_in[2];
    const float* b1        = (const float*)d_in[3];
    const float* W2        = (const float*)d_in[4];
    const float* Wt        = (const float*)d_in[5];
    const float* bt        = (const float*)d_in[6];
    const float* Wp        = (const float*)d_in[7];
    const float* bp        = (const float*)d_in[8];
    const float* w_pos     = (const float*)d_in[9];
    const float* b_pos     = (const float*)d_in[10];
    const float* alpha     = (const float*)d_in[11];

    float* out    = (float*)d_out;
    float* z_out  = out + OFF_Z;
    float* tl_out = out + OFF_TL;
    float* pl_out = out + OFF_PL;
    float* tc_out = out + OFF_TC;
    float* cv_out = out + OFF_CV;

    char*      ws     = (char*)d_ws;
    _Float16*  h16    = (_Float16*)(ws + WS_H16);
    _Float16*  z16    = (_Float16*)(ws + WS_Z16);
    _Float16*  w1s    = (_Float16*)(ws + WS_W1S);
    _Float16*  w2s    = (_Float16*)(ws + WS_W2S);
    _Float16*  wts    = (_Float16*)(ws + WS_WTS);
    _Float16*  wps    = (_Float16*)(ws + WS_WPS);
    float*     dotpos = (float*)   (ws + WS_DOT);
    int*       maxpos = (int*)     (ws + WS_MAXPOS);

    dim3 blk(256);

    // 0) pre-swizzle all weights into WMMA B-fragment layout (f16, padded)
    {
        int t1 = (IN_DIM / 32) * NT_W1 * 32;   // 81920
        int t2 = (BELLY / 32) * NT_W2 * 32;    // 32768
        int t3 = (SCALE / 32) * NT_WT * 32;    // 4096
        int t4 = (SCALE / 32) * NT_WP * 32;    // 32768
        swizzle_w_kernel<<<dim3((t1 + 255) / 256), blk, 0, stream>>>(W1, w1s, IN_DIM, BELLY, NT_W1);
        swizzle_w_kernel<<<dim3((t2 + 255) / 256), blk, 0, stream>>>(W2, w2s, BELLY, SCALE, NT_W2);
        swizzle_w_kernel<<<dim3((t3 + 255) / 256), blk, 0, stream>>>(Wt, wts, SCALE, NBINS, NT_WT);
        swizzle_w_kernel<<<dim3((t4 + 255) / 256), blk, 0, stream>>>(Wp, wps, SCALE, NBINS * NPAT, NT_WP);
    }

    // 1) h = relu(features @ W1 + b1) -> f16 in ws   (A: f32 global, cvt in staging)
    gemm_wmma_kernel<1, false, true><<<dim3(NT_W1 / 8, BATCH / BM), blk, 0, stream>>>(
        (const void*)features, w1s, b1, (void*)h16, BATCH, BELLY, NT_W1, IN_DIM);

    // 2) z_raw = h @ W2 -> f32 into d_out z segment
    gemm_wmma_kernel<0, true, false><<<dim3(NT_W2 / 8, BATCH / BM), blk, 0, stream>>>(
        (const void*)h16, w2s, nullptr, (void*)z_out, BATCH, SCALE, NT_W2, BELLY);

    // 3) row-normalize z (in place) + f16 copy + dot(z, w_pos)
    normalize_kernel<<<dim3(BATCH / 8), blk, 0, stream>>>(z_out, z16, dotpos, w_pos);

    // 4) timestep_class + max_pos
    zero_maxpos_kernel<<<dim3(1), dim3(1), 0, stream>>>(maxpos);
    tclass_kernel<<<dim3((BATCH + 255) / 256), blk, 0, stream>>>(timesteps, tc_out, maxpos);

    // 5) heads
    gemm_wmma_kernel<0, true, false><<<dim3(NT_WT / 8, BATCH / BM), blk, 0, stream>>>(
        (const void*)z16, wts, bt, (void*)tl_out, BATCH, NBINS, NT_WT, SCALE);
    gemm_wmma_kernel<0, true, false><<<dim3(NT_WP / 8, BATCH / BM), blk, 0, stream>>>(
        (const void*)z16, wps, bp, (void*)pl_out, BATCH, NBINS * NPAT, NT_WP, SCALE);

    // 6) cantor staircase
    cantor_kernel<<<dim3((BATCH + 255) / 256), blk, 0, stream>>>(
        timesteps, dotpos, b_pos, alpha, maxpos, cv_out);
}
